// Attention_30408368455750
// MI455X (gfx1250) — matrix operand order repaired and checked by reading, not compiled, not run on hardware
//
#include <hip/hip_runtime.h>
#include <hip/hip_bf16.h>

typedef __attribute__((ext_vector_type(16))) _Float16 v16h;
typedef __attribute__((ext_vector_type(8)))  _Float16 v8h;
typedef __attribute__((ext_vector_type(8)))  float    v8f;
typedef __attribute__((ext_vector_type(4)))  float    v4f;

#define HIDDEN 1024
#define HEADS  16
#define HEAD   64
#define SEQ    2048
#define BATCH  4
#define M_TOTAL (BATCH*SEQ)      // 8192 rows for the projection GEMMs

// ---------------------------------------------------------------------------
// Kernel 1a: convert x fp32 -> f16 (once; keeps cvt chains out of the GEMM)
// ---------------------------------------------------------------------------
__global__ __launch_bounds__(256) void cvt_x_kernel(const float* __restrict__ x,
                                                    _Float16* __restrict__ xh) {
    size_t i = ((size_t)blockIdx.x * 256 + threadIdx.x) * 4;
    v4f v = *(const v4f*)(x + i);
    xh[i + 0] = (_Float16)v.x;
    xh[i + 1] = (_Float16)v.y;
    xh[i + 2] = (_Float16)v.z;
    xh[i + 3] = (_Float16)v.w;
}

// ---------------------------------------------------------------------------
// Kernel 1b: convert Wq/Wk/Wv fp32 -> f16
// ---------------------------------------------------------------------------
__global__ __launch_bounds__(256) void cvt_w_kernel(const float* __restrict__ Wq,
                                                    const float* __restrict__ Wk,
                                                    const float* __restrict__ Wv,
                                                    _Float16* __restrict__ Wh) {
    const size_t N = (size_t)HIDDEN * HIDDEN;
    size_t i = (size_t)blockIdx.x * blockDim.x + threadIdx.x;
    Wh[i]         = (_Float16)Wq[i];
    Wh[N + i]     = (_Float16)Wk[i];
    Wh[2 * N + i] = (_Float16)Wv[i];
}

// ---------------------------------------------------------------------------
// Kernel 2: QKV projections.
// Block = 128 threads (4 waves). Block tile = 32 rows x 256 cols; each wave
// owns 32x64 (8 WMMA accumulators). The 32x32 f16 A-panel per K-step is
// staged into LDS with GLOBAL_LOAD_ASYNC_TO_LDS_B128 (ASYNCcnt), DOUBLE
// BUFFERED so the chunk-(i+1) DMA overlaps the chunk-i WMMA block.
// B comes straight from W rows (W[N,K] row-major == B^T): 16 contiguous
// halves per lane -> one 32B global load per fragment.
// Stores: Q,K f16 [b,h,s,d]; V transposed f16 [b,h,d,s].
// ---------------------------------------------------------------------------
__global__ __launch_bounds__(128) void qkv_gemm_kernel(const _Float16* __restrict__ xh,
                                                       const _Float16* __restrict__ Wh,
                                                       const float* __restrict__ bq,
                                                       const float* __restrict__ bk,
                                                       const float* __restrict__ bvv,
                                                       _Float16* __restrict__ qh,
                                                       _Float16* __restrict__ kh,
                                                       _Float16* __restrict__ vt) {
    __shared__ _Float16 Xs[2][32 * 32];

    const int t    = threadIdx.x;
    const int lane = t & 31;
    const int wave = t >> 5;
    const int hi   = lane >> 4;
    const int l15  = lane & 15;

    const int bx    = blockIdx.x;        // 3 * 256 * 4 = 3072 blocks
    const int which = bx >> 10;          // 0=q 1=k 2=v
    const int rem   = bx & 1023;
    const int mt32  = rem >> 2;          // 0..255 : 32-row block
    const int cb    = rem & 3;           // 0..3   : 256-col block
    const int nbase = cb * 256 + wave * 64;

    const _Float16* Wbase = Wh + (size_t)which * HIDDEN * HIDDEN;
    const _Float16* Wr0 = Wbase + (size_t)(nbase +  0 + l15) * HIDDEN;
    const _Float16* Wr1 = Wbase + (size_t)(nbase + 16 + l15) * HIDDEN;
    const _Float16* Wr2 = Wbase + (size_t)(nbase + 32 + l15) * HIDDEN;
    const _Float16* Wr3 = Wbase + (size_t)(nbase + 48 + l15) * HIDDEN;

    // async A-panel staging: thread t moves 16B (8 halves) of row (t>>2)
    const int xrow = t >> 2;
    const int xcol = (t & 3) * 8;
    unsigned long long gaddr =
        (unsigned long long)(xh + (size_t)(mt32 * 32 + xrow) * HIDDEN + xcol);
    unsigned ldsoff0 = (unsigned)(size_t)(&Xs[0][xrow * 32 + xcol]);
    unsigned ldsoff1 = (unsigned)(size_t)(&Xs[1][xrow * 32 + xcol]);

    v8f acc[4][2] = {};

    // prologue: stage chunk 0 into buffer 0
    asm volatile("global_load_async_to_lds_b128 %0, %1, off"
                 :: "v"(ldsoff0), "v"(gaddr) : "memory");

    for (int i = 0; i < HIDDEN / 32; ++i) {
        const int kb = i * 32;
        // chunk i is the only outstanding async op -> wait it, make visible
        asm volatile("s_wait_asynccnt 0x0" ::: "memory");
        __syncthreads();

        // overlap: kick off chunk i+1 into the other buffer now
        if (i + 1 < HIDDEN / 32) {
            asm volatile("global_load_async_to_lds_b128 %0, %1, off"
                         :: "v"((i & 1) ? ldsoff0 : ldsoff1),
                            "v"(gaddr + (unsigned long long)(kb + 32) * 2)
                         : "memory");
        }

        const _Float16* Xc = Xs[i & 1];

        // ---- A fragments (rows 0-15 and 16-31) from LDS ----
        v16h a0, a1;
        {
            v8h p0 = *(const v8h*)(&Xc[l15 * 32 + hi * 16]);
            v8h p1 = *(const v8h*)(&Xc[l15 * 32 + hi * 16 + 8]);
            v8h p2 = *(const v8h*)(&Xc[(16 + l15) * 32 + hi * 16]);
            v8h p3 = *(const v8h*)(&Xc[(16 + l15) * 32 + hi * 16 + 8]);
#pragma unroll
            for (int e = 0; e < 8; ++e) {
                a0[e] = p0[e]; a0[8 + e] = p1[e];
                a1[e] = p2[e]; a1[8 + e] = p3[e];
            }
        }

        // ---- 4 B fragments, 8 WMMAs ----
        v16h b0 = *(const v16h*)(Wr0 + kb + hi * 16);
        v16h b1 = *(const v16h*)(Wr1 + kb + hi * 16);
        v16h b2 = *(const v16h*)(Wr2 + kb + hi * 16);
        v16h b3 = *(const v16h*)(Wr3 + kb + hi * 16);
        acc[0][0] = __builtin_amdgcn_wmma_f32_16x16x32_f16(false, a0, false, b0, (short)0, acc[0][0], false, false);
        acc[0][1] = __builtin_amdgcn_wmma_f32_16x16x32_f16(false, a1, false, b0, (short)0, acc[0][1], false, false);
        acc[1][0] = __builtin_amdgcn_wmma_f32_16x16x32_f16(false, a0, false, b1, (short)0, acc[1][0], false, false);
        acc[1][1] = __builtin_amdgcn_wmma_f32_16x16x32_f16(false, a1, false, b1, (short)0, acc[1][1], false, false);
        acc[2][0] = __builtin_amdgcn_wmma_f32_16x16x32_f16(false, a0, false, b2, (short)0, acc[2][0], false, false);
        acc[2][1] = __builtin_amdgcn_wmma_f32_16x16x32_f16(false, a1, false, b2, (short)0, acc[2][1], false, false);
        acc[3][0] = __builtin_amdgcn_wmma_f32_16x16x32_f16(false, a0, false, b3, (short)0, acc[3][0], false, false);
        acc[3][1] = __builtin_amdgcn_wmma_f32_16x16x32_f16(false, a1, false, b3, (short)0, acc[3][1], false, false);
        // no trailing barrier: next overwrite of this buffer happens only
        // after the next iteration's barrier, and our LDS reads were
        // dscnt-waited before the WMMAs above.
    }

    // ---- epilogue: bias add, f16 convert, scatter to q/k/v layouts ----
    const float* bias = (which == 0) ? bq : (which == 1) ? bk : bvv;
#pragma unroll
    for (int j = 0; j < 4; ++j) {
        int   nglob = nbase + j * 16 + l15;
        float bn    = bias[nglob];
        int   h     = nglob >> 6;
        int   d     = nglob & 63;
#pragma unroll
        for (int half = 0; half < 2; ++half) {
#pragma unroll
            for (int r = 0; r < 8; ++r) {
                int m  = mt32 * 32 + half * 16 + r + hi * 8;
                int bz = m >> 11;
                int s  = m & 2047;
                _Float16 val = (_Float16)(acc[j][half][r] + bn);
                size_t bh = (size_t)bz * HEADS + h;
                if (which == 0)      qh[(bh * SEQ + s) * HEAD + d] = val;
                else if (which == 1) kh[(bh * SEQ + s) * HEAD + d] = val;
                else                 vt[(bh * HEAD + d) * SEQ + s] = val;
            }
        }
    }
}

// ---------------------------------------------------------------------------
// Kernel 3: flash attention. One wave == one (b,h,16-query) tile.
// Online softmax over 128 key tiles, 2 tiles (32 keys) per step:
//   4 WMMA for Q@K^T (K-dim = d = 64), softmax update, P transposed via LDS,
//   4 WMMA for P@V (accumulating 16x64 O tile).
// ---------------------------------------------------------------------------
__global__ __launch_bounds__(128) void attn_kernel(const _Float16* __restrict__ qh,
                                                   const _Float16* __restrict__ kh,
                                                   const _Float16* __restrict__ vt,
                                                   float* __restrict__ out) {
    __shared__ _Float16 Pl[4][16 * 32];

    const int lane = threadIdx.x & 31;
    const int wave = threadIdx.x >> 5;
    const int hi   = lane >> 4;
    const int l15  = lane & 15;

    int idx = blockIdx.x * 4 + wave;   // 8192 total = 64 bh * 128 qtiles
    int qt  = idx & 127;
    int bh  = idx >> 7;

    const _Float16* Q = qh + ((size_t)bh * SEQ + qt * 16 + l15) * HEAD;
    v16h aq0, aq1;
    {
        v8h a0 = *(const v8h*)(Q + hi * 16);
        v8h a1 = *(const v8h*)(Q + hi * 16 + 8);
        v8h a2 = *(const v8h*)(Q + 32 + hi * 16);
        v8h a3 = *(const v8h*)(Q + 32 + hi * 16 + 8);
#pragma unroll
        for (int e = 0; e < 8; ++e) {
            aq0[e] = a0[e]; aq0[8 + e] = a1[e];
            aq1[e] = a2[e]; aq1[8 + e] = a3[e];
        }
    }

    float m_run[8], l_run[8];
#pragma unroll
    for (int r = 0; r < 8; ++r) { m_run[r] = -1e30f; l_run[r] = 0.0f; }
    v8f o0 = {}, o1 = {}, o2 = {}, o3 = {};

    _Float16* P = Pl[wave];
    const _Float16* Kb = kh + (size_t)bh * SEQ * HEAD;
    const _Float16* Vb = vt + (size_t)bh * HEAD * SEQ;

    for (int j2 = 0; j2 < SEQ / 16; j2 += 2) {
        // ---- scores: two 16x16 tiles ----
        v8f s0 = {}, s1 = {};
        {
            const _Float16* kr0 = Kb + (size_t)(j2 * 16 + l15) * HEAD;
            v16h bk0 = *(const v16h*)(kr0 + hi * 16);
            v16h bk1 = *(const v16h*)(kr0 + 32 + hi * 16);
            s0 = __builtin_amdgcn_wmma_f32_16x16x32_f16(false, aq0, false, bk0, (short)0, s0, false, false);
            s0 = __builtin_amdgcn_wmma_f32_16x16x32_f16(false, aq1, false, bk1, (short)0, s0, false, false);
            const _Float16* kr1 = Kb + (size_t)((j2 + 1) * 16 + l15) * HEAD;
            v16h bk2 = *(const v16h*)(kr1 + hi * 16);
            v16h bk3 = *(const v16h*)(kr1 + 32 + hi * 16);
            s1 = __builtin_amdgcn_wmma_f32_16x16x32_f16(false, aq0, false, bk2, (short)0, s1, false, false);
            s1 = __builtin_amdgcn_wmma_f32_16x16x32_f16(false, aq1, false, bk3, (short)0, s1, false, false);
        }

        // ---- online softmax update (C layout: lane = key col, vgpr r = row) ----
#pragma unroll
        for (int r = 0; r < 8; ++r) {
            float a = s0[r] * 0.125f;   // 1/sqrt(64)
            float b = s1[r] * 0.125f;
            float rm = fmaxf(a, b);
            rm = fmaxf(rm, __shfl_xor(rm, 1, 16));
            rm = fmaxf(rm, __shfl_xor(rm, 2, 16));
            rm = fmaxf(rm, __shfl_xor(rm, 4, 16));
            rm = fmaxf(rm, __shfl_xor(rm, 8, 16));
            float mn = fmaxf(m_run[r], rm);
            float sc = __expf(m_run[r] - mn);
            m_run[r] = mn;
            float p0 = __expf(a - mn);
            float p1 = __expf(b - mn);
            float rs = p0 + p1;
            rs += __shfl_xor(rs, 1, 16);
            rs += __shfl_xor(rs, 2, 16);
            rs += __shfl_xor(rs, 4, 16);
            rs += __shfl_xor(rs, 8, 16);
            l_run[r] = l_run[r] * sc + rs;
            o0[r] *= sc; o1[r] *= sc; o2[r] *= sc; o3[r] *= sc;
            P[(r + hi * 8) * 32 + l15]      = (_Float16)p0;
            P[(r + hi * 8) * 32 + 16 + l15] = (_Float16)p1;
        }

        // ---- P in A-fragment layout (transpose through LDS) ----
        v16h ap;
        {
            v8h q0 = *(const v8h*)(P + l15 * 32 + hi * 16);
            v8h q1 = *(const v8h*)(P + l15 * 32 + hi * 16 + 8);
#pragma unroll
            for (int e = 0; e < 8; ++e) { ap[e] = q0[e]; ap[8 + e] = q1[e]; }
        }

        // ---- P @ V over 4 d-chunks ----
        const _Float16* vr = Vb + (size_t)l15 * SEQ + j2 * 16 + hi * 16;
        v16h bv0 = *(const v16h*)(vr);
        v16h bv1 = *(const v16h*)(vr + 16 * SEQ);
        v16h bv2 = *(const v16h*)(vr + 32 * SEQ);
        v16h bv3 = *(const v16h*)(vr + 48 * SEQ);
        o0 = __builtin_amdgcn_wmma_f32_16x16x32_f16(false, ap, false, bv0, (short)0, o0, false, false);
        o1 = __builtin_amdgcn_wmma_f32_16x16x32_f16(false, ap, false, bv1, (short)0, o1, false, false);
        o2 = __builtin_amdgcn_wmma_f32_16x16x32_f16(false, ap, false, bv2, (short)0, o2, false, false);
        o3 = __builtin_amdgcn_wmma_f32_16x16x32_f16(false, ap, false, bv3, (short)0, o3, false, false);
    }

    const int bz = bh >> 4;
    const int h  = bh & 15;
#pragma unroll
    for (int r = 0; r < 8; ++r) {
        float inv = 1.0f / l_run[r];
        size_t row = (size_t)(bz * SEQ + qt * 16 + r + hi * 8) * HIDDEN + h * HEAD;
        out[row +      l15] = o0[r] * inv;
        out[row + 16 + l15] = o1[r] * inv;
        out[row + 32 + l15] = o2[r] * inv;
        out[row + 48 + l15] = o3[r] * inv;
    }
}

// ---------------------------------------------------------------------------
extern "C" void kernel_launch(void* const* d_in, const int* in_sizes, int n_in,
                              void* d_out, int out_size, void* d_ws, size_t ws_size,
                              hipStream_t stream) {
    (void)in_sizes; (void)n_in; (void)out_size; (void)ws_size;
    const float* x  = (const float*)d_in[0];
    const float* Wq = (const float*)d_in[1];
    const float* bq = (const float*)d_in[2];
    const float* Wk = (const float*)d_in[3];
    const float* bk = (const float*)d_in[4];
    const float* Wv = (const float*)d_in[5];
    const float* bv = (const float*)d_in[6];
    float* out = (float*)d_out;

    // workspace layout (halves): xh | Wh[3] | qh | kh | vt  (~73.4 MB)
    const size_t XN  = (size_t)M_TOTAL * HIDDEN;   // 8,388,608
    const size_t WN  = (size_t)HIDDEN * HIDDEN;    // 1,048,576
    _Float16* xh = (_Float16*)d_ws;
    _Float16* Wh = xh + XN;
    _Float16* qh = Wh + 3 * WN;
    _Float16* kh = qh + XN;
    _Float16* vt = kh + XN;

    // 1) fp32 -> f16 conversions
    cvt_x_kernel<<<XN / (256 * 4), 256, 0, stream>>>(x, xh);
    cvt_w_kernel<<<WN / 256, 256, 0, stream>>>(Wq, Wk, Wv, Wh);

    // 2) QKV projections: 3 * 256 row-blocks * 4 col-blocks
    qkv_gemm_kernel<<<3 * 256 * 4, 128, 0, stream>>>(xh, Wh, bq, bk, bv,
                                                     qh, kh, vt);

    // 3) attention: 64 bh * 128 qtiles = 8192 waves, 4 waves/block
    attn_kernel<<<8192 / 4, 128, 0, stream>>>(qh, kh, vt, out);
}
